// InvariantCostVolume_86663850098775
// MI455X (gfx1250) — matrix operand order repaired and checked by reading, not compiled, not run on hardware
//
#include <hip/hip_runtime.h>
#include <hip/hip_bf16.h>

// ---------------------------------------------------------------------------
// CDNA5 (gfx1250, wave32) implementation of InvariantCostVolume.
// Heavy work = two 3x3 convs as implicit GEMM on v_wmma_f32_16x16x32_bf16,
// async global->LDS weight staging, double-buffered LDS, full-row M tiles.
// ---------------------------------------------------------------------------

typedef __attribute__((ext_vector_type(16))) __bf16 v16bf;
typedef __attribute__((ext_vector_type(8)))  float  v8f;
typedef int v4i_vs __attribute__((vector_size(16)));   // matches builtin's V4i

#if defined(__has_builtin)
# if __has_builtin(__builtin_amdgcn_global_load_async_to_lds_b128) && \
     __has_builtin(__builtin_amdgcn_s_wait_asynccnt)
#  define HAVE_ASYNC_LDS 1
# endif
#endif
#ifndef HAVE_ASYNC_LDS
# define HAVE_ASYNC_LDS 0
#endif

#if HAVE_ASYNC_LDS
typedef __attribute__((address_space(1))) v4i_vs* gv4p_t;   // global v4i*
typedef __attribute__((address_space(3))) v4i_vs* lv4p_t;   // LDS v4i*
#endif

__device__ __forceinline__ void async_copy_b128(const void* g, void* l) {
#if HAVE_ASYNC_LDS
    __builtin_amdgcn_global_load_async_to_lds_b128((gv4p_t)g, (lv4p_t)l, 0, 0);
#else
    *(uint4*)l = *(const uint4*)g;
#endif
}
__device__ __forceinline__ void async_wait_all() {
#if HAVE_ASYNC_LDS
    __builtin_amdgcn_s_wait_asynccnt(0);
#endif
}

__device__ __forceinline__ unsigned short f2bf(float f) {
    unsigned u = __float_as_uint(f);
    u += 0x7FFFu + ((u >> 16) & 1u);          // round-to-nearest-even
    return (unsigned short)(u >> 16);
}
__device__ __forceinline__ float bf2f(unsigned short h) {
    return __uint_as_float(((unsigned)h) << 16);
}

// ---------------------------------------------------------------------------
// Weight packing: W[oc][ic][3][3] f32 -> Wp[kk][oc][ic] bf16.
// Rows are contiguous in ic (=K), so B tiles stage as dense 16B chunks and a
// WMMA B fragment is a single contiguous 32B per-lane LDS read.
// ---------------------------------------------------------------------------
__global__ void pack_w_kernel(const float* __restrict__ w,
                              unsigned short* __restrict__ wp,
                              int OC, int IC) {
    int idx = blockIdx.x * blockDim.x + threadIdx.x;
    int total = OC * IC * 9;
    if (idx >= total) return;
    int ic = idx % IC;
    int r  = idx / IC;
    int oc = r % OC;
    int kk = r / OC;
    wp[idx] = f2bf(w[((size_t)oc * IC + ic) * 9 + kk]);
}

// R1 (32,64,3,3,3): conv3d input is cost_agg broadcast over all 64 channels and
// only kw=1 is in bounds -> reduce to R1r[o][kd*3+kh] = sum_c R1[o,c,kd,kh,1]
__global__ void prep_r1_kernel(const float* __restrict__ R1, float* __restrict__ R1r) {
    int idx = blockIdx.x * blockDim.x + threadIdx.x;
    if (idx >= 32 * 9) return;
    int o = idx / 9, kk = idx % 9;
    int kd = kk / 3, kh = kk % 3;
    float s = 0.f;
    for (int c = 0; c < 64; ++c)
        s += R1[((size_t)(o * 64 + c)) * 27 + kd * 9 + kh * 3 + 1];
    R1r[idx] = s;
}

// R2 (16,32,3,3,3): keep only kw=1 slice -> R2r[o][c][kd*3+kh]
__global__ void prep_r2_kernel(const float* __restrict__ R2, float* __restrict__ R2r) {
    int idx = blockIdx.x * blockDim.x + threadIdx.x;
    if (idx >= 16 * 32 * 9) return;
    int kk = idx % 9;
    int c  = (idx / 9) % 32;
    int o  = idx / (9 * 32);
    int kd = kk / 3, kh = kk % 3;
    R2r[idx] = R2[((size_t)(o * 32 + c)) * 27 + kd * 9 + kh * 3 + 1];
}

// ---------------------------------------------------------------------------
// Implicit-GEMM 3x3 conv + bias + ReLU on WMMA bf16, double-buffered LDS.
//   Block = 128 threads = 4 wave32.  Tile: M=96 pixels (full row) x N=64 oc.
//   A tile: bf16, pre-swizzled to fragment layout k = 16*(lane>=16) + e, same as B (6 frags).
//   B tile: Braw[oc][k] rows staged via GLOBAL_LOAD_ASYNC_TO_LDS_B128; a lane's
//           fragment is the contiguous 32B at (row = n, half = 16*(lane>=16)).
//   All fragments are hoisted into registers, then 6 WMMAs issue back-to-back.
// ---------------------------------------------------------------------------
template<int IC, int OCSEGS, bool PAIR>
__global__ __launch_bounds__(128)
void conv3x3_wmma(const float* __restrict__ fin_f32,
                  const unsigned short* __restrict__ fin_bf,
                  const unsigned short* __restrict__ wp,   // [9][OC][IC] bf16
                  const float* __restrict__ bias,
                  unsigned short* __restrict__ out) {      // [10][OC][96][96] bf16
    constexpr int OC    = OCSEGS * 64;
    constexpr int KCH   = IC / 32;          // 32-channel chunks per tap
    constexpr int NSTEP = 9 * KCH;

    int bid   = blockIdx.x;
    const int ocseg = bid % OCSEGS;  bid /= OCSEGS;
    const int y     = bid % 96;
    const int img   = bid / 96;      // 0..9  (b*5+n)

    const int t    = threadIdx.x;
    const int lane = t & 31;
    const int wave = t >> 5;
    const int oc0  = ocseg * 64;

    __shared__ __align__(64) unsigned short At[2][6][32][16];   // [buf][fragM]
    __shared__ __align__(64) unsigned short Braw[2][64][32];    // [buf][oc][k]

    v8f acc[6];
    #pragma unroll
    for (int f = 0; f < 6; ++f) acc[f] = (v8f){};

    // per-thread constants of the A staging swizzle
    // A layout (matches B): k = e + (lane>=16?16:0)
    const int kA        = t >> 2;             // local K channel staged by t
    const int px0       = (t & 3) * 24;       // first of 24 pixels staged by t
    const int laneHalfA = (kA & 16) ? 16 : 0;
    const int eA        = kA & 15;

    auto fill = [&](int ks, int buf) {
        const int kk  = ks / KCH;
        const int icc = (ks % KCH) * 32;
        const int ky = kk / 3 - 1, kx = kk % 3 - 1;
        const int yy  = y + ky;
        const bool yok = (unsigned)yy < 96u;
        const int yc  = yy < 0 ? 0 : (yy > 95 ? 95 : yy);

        // ---- B tile: 64 rows x 64B (contig in k), async global->LDS ----
        {
            #pragma unroll
            for (int j = 0; j < 2; ++j) {
                const int q  = t * 2 + j;        // 256 x 16B chunks
                const int oc = q >> 2;
                const int co = (q & 3) * 8;      // bf16 elements within row
                async_copy_b128(wp + ((size_t)kk * OC + oc0 + oc) * IC + icc + co,
                                &Braw[buf][oc][co]);
            }
            if (ks + 2 < NSTEP) {                // global_prefetch_b8
                const int kk2  = (ks + 2) / KCH;
                const int icc2 = ((ks + 2) % KCH) * 32;
                __builtin_prefetch(wp + ((size_t)kk2 * OC + oc0) * IC + icc2, 0, 1);
            }
        }
        // ---- A tile: clamped unconditional loads, masked, swizzled scatter ----
        {
            const int c2 = icc + kA;
            const float*          srcF = nullptr;
            const unsigned short* srcH = nullptr;
            if (PAIR) {
                const int b    = img / 5;
                const int simg = (c2 < IC / 2) ? img : b * 5;  // feat_ref = n==0
                const int c    = (c2 < IC / 2) ? c2 : c2 - IC / 2;
                srcF = fin_f32 + ((size_t)simg * (IC / 2) + c) * 9216 + yc * 96;
            } else {
                srcH = fin_bf + ((size_t)img * IC + c2) * 9216 + yc * 96;
            }
            float          fv[24];
            unsigned short hv[24];
            bool           ok[24];
            #pragma unroll
            for (int j = 0; j < 24; ++j) {       // all loads issue, no exec mask
                const int xx = px0 + j + kx;
                const int xc = xx < 0 ? 0 : (xx > 95 ? 95 : xx);
                ok[j] = yok && (unsigned)xx < 96u;
                if (PAIR) fv[j] = srcF[xc]; else hv[j] = srcH[xc];
            }
            #pragma unroll
            for (int j = 0; j < 24; ++j) {
                unsigned short h = PAIR ? f2bf(fv[j]) : hv[j];
                h = ok[j] ? h : (unsigned short)0;
                const int m = px0 + j;
                At[buf][m >> 4][(m & 15) + laneHalfA][eA] = h;
            }
        }
    };

    fill(0, 0);
    async_wait_all();
    __syncthreads();

    int buf = 0;
    for (int ks = 0; ks < NSTEP; ++ks) {
        if (ks + 1 < NSTEP) fill(ks + 1, buf ^ 1);   // overlap next-tile staging

        // ---- hoist all fragments, then issue the WMMA chain ----
        // B fragment: lane holds n = lane&15 (row), k = 16*(lane>=16) + e
        const int brow = (wave << 4) + (lane & 15);
        const int bkh  = (lane >> 4) << 4;
        const v16bf bw = *reinterpret_cast<const v16bf*>(&Braw[buf][brow][bkh]);
        v16bf a[6];
        #pragma unroll
        for (int f = 0; f < 6; ++f)
            a[f] = *reinterpret_cast<const v16bf*>(&At[buf][f][lane][0]);

        #pragma unroll
        for (int f = 0; f < 6; ++f)
            acc[f] = __builtin_amdgcn_wmma_f32_16x16x32_bf16(
                         false, a[f], false, bw, (short)0, acc[f], false, false);

        async_wait_all();        // my async copies into buf^1 landed in LDS
        __syncthreads();         // everyone done staging buf^1 / reading buf
        buf ^= 1;
    }

    // ----- epilogue: bias + ReLU, store bf16 NCHW -----
    const int ocw = oc0 + wave * 16 + (lane & 15);          // D: n = lane&15
    const float bsv = bias[ocw];
    unsigned short* obase = out + ((size_t)img * OC + ocw) * 9216 + y * 96;
    #pragma unroll
    for (int f = 0; f < 6; ++f) {
        #pragma unroll
        for (int r = 0; r < 8; ++r) {
            const int m = r + ((lane >> 4) << 3);           // D: m = r + 8*(lane/16)
            float v0 = acc[f][r] + bsv;
            v0 = v0 > 0.f ? v0 : 0.f;
            obase[f * 16 + m] = f2bf(v0);
        }
    }
}

// ---------------------------------------------------------------------------
// 1x1 conv (128->1) + mean over N=5  ->  cost_agg[b][y][x]
// ---------------------------------------------------------------------------
__global__ void cost_mean_kernel(const unsigned short* __restrict__ O2,
                                 const float* __restrict__ W3,
                                 const float* __restrict__ b3,
                                 float* __restrict__ costagg) {
    int idx = blockIdx.x * blockDim.x + threadIdx.x;
    if (idx >= 2 * 9216) return;
    const int b = idx / 9216, p = idx % 9216;
    float acc = 0.f;
    for (int n = 0; n < 5; ++n) {
        const unsigned short* src = O2 + ((size_t)(b * 5 + n) * 128) * 9216 + p;
        for (int c = 0; c < 128; ++c)
            acc += W3[c] * bf2f(src[(size_t)c * 9216]);
    }
    costagg[idx] = acc * 0.2f + b3[0];
}

// collapsed conv3d #1: 1 -> 32 channels, 3x3 over (d,h) plane
__global__ void v1_kernel(const float* __restrict__ costagg,
                          const float* __restrict__ R1r,
                          const float* __restrict__ rb1,
                          float* __restrict__ v1b) {
    int idx = blockIdx.x * blockDim.x + threadIdx.x;
    if (idx >= 2 * 32 * 9216) return;
    const int x = idx % 96;
    const int y = (idx / 96) % 96;
    const int o = (idx / 9216) % 32;
    const int b = idx / (32 * 9216);
    float a = rb1[o];
    #pragma unroll
    for (int kk = 0; kk < 9; ++kk) {
        const int yy = y + kk / 3 - 1, xx = x + kk % 3 - 1;
        if ((unsigned)yy < 96u && (unsigned)xx < 96u)
            a += R1r[o * 9 + kk] * costagg[b * 9216 + yy * 96 + xx];
    }
    v1b[idx] = a > 0.f ? a : 0.f;
}

// collapsed conv3d #2: 32 -> 16 channels, 3x3
__global__ void v2_kernel(const float* __restrict__ v1b,
                          const float* __restrict__ R2r,
                          const float* __restrict__ rb2,
                          float* __restrict__ v2b) {
    int idx = blockIdx.x * blockDim.x + threadIdx.x;
    if (idx >= 2 * 16 * 9216) return;
    const int x = idx % 96;
    const int y = (idx / 96) % 96;
    const int o = (idx / 9216) % 16;
    const int b = idx / (16 * 9216);
    float a = rb2[o];
    for (int c = 0; c < 32; ++c) {
        const float* src = v1b + ((size_t)(b * 32 + c)) * 9216;
        const float* wr  = R2r + (o * 32 + c) * 9;
        #pragma unroll
        for (int kk = 0; kk < 9; ++kk) {
            const int yy = y + kk / 3 - 1, xx = x + kk % 3 - 1;
            if ((unsigned)yy < 96u && (unsigned)xx < 96u)
                a += wr[kk] * src[yy * 96 + xx];
        }
    }
    v2b[idx] = a > 0.f ? a : 0.f;
}

// 1x1 16->64 + softmax(-cv) over the 64 bins; one wave32 per pixel,
// 2 bins per lane, __shfl_xor reductions. Writes NHWC cost_volume then probs.
__global__ __launch_bounds__(128)
void softmax_kernel(const float* __restrict__ v2b,
                    const float* __restrict__ R3,
                    const float* __restrict__ rb3,
                    float* __restrict__ outp) {
    const int wave = threadIdx.x >> 5;
    const int lane = threadIdx.x & 31;
    const int p = blockIdx.x * 4 + wave;        // 0..18431
    const int b = p / 9216, rem = p % 9216;

    float vv[16];
    const float* src = v2b + (size_t)b * 16 * 9216 + rem;
    #pragma unroll
    for (int c = 0; c < 16; ++c) vv[c] = src[(size_t)c * 9216];

    const int o0 = lane, o1 = lane + 32;
    float cv0 = rb3[o0], cv1 = rb3[o1];
    #pragma unroll
    for (int c = 0; c < 16; ++c) {
        cv0 += R3[o0 * 16 + c] * vv[c];
        cv1 += R3[o1 * 16 + c] * vv[c];
    }
    float t0 = -cv0, t1 = -cv1;
    float mx = fmaxf(t0, t1);
    for (int off = 16; off; off >>= 1) mx = fmaxf(mx, __shfl_xor(mx, off, 32));
    const float e0 = __expf(t0 - mx), e1 = __expf(t1 - mx);
    float s = e0 + e1;
    for (int off = 16; off; off >>= 1) s += __shfl_xor(s, off, 32);
    const float inv = 1.f / s;

    const size_t base = (size_t)p * 64;
    const size_t off2 = (size_t)2 * 9216 * 64;
    outp[base + o0] = cv0;
    outp[base + o1] = cv1;
    outp[off2 + base + o0] = e0 * inv;
    outp[off2 + base + o1] = e1 * inv;
}

// ---------------------------------------------------------------------------
extern "C" void kernel_launch(void* const* d_in, const int* in_sizes, int n_in,
                              void* d_out, int out_size, void* d_ws, size_t ws_size,
                              hipStream_t stream) {
    (void)in_sizes; (void)n_in; (void)out_size; (void)ws_size;

    const float* focal = (const float*)d_in[0];
    const float* W1  = (const float*)d_in[3];
    const float* b1  = (const float*)d_in[4];
    const float* W2  = (const float*)d_in[5];
    const float* b2  = (const float*)d_in[6];
    const float* W3  = (const float*)d_in[7];
    const float* b3  = (const float*)d_in[8];
    const float* R1  = (const float*)d_in[9];
    const float* rb1 = (const float*)d_in[10];
    const float* R2  = (const float*)d_in[11];
    const float* rb2 = (const float*)d_in[12];
    const float* R3  = (const float*)d_in[13];
    const float* rb3 = (const float*)d_in[14];

    char* ws = (char*)d_ws;
    unsigned short* Wp1 = (unsigned short*)(ws + 0);          //  2,359,296 B
    unsigned short* Wp2 = (unsigned short*)(ws + 2359296);    //    589,824 B
    unsigned short* O1  = (unsigned short*)(ws + 2949120);    // 47,185,920 B
    unsigned short* O2  = (unsigned short*)(ws + 50135040);   // 23,592,960 B
    float* costagg = (float*)(ws + 73728000);                 //     73,728 B
    float* R1r     = (float*)(ws + 73801728);                 //      1,152 B
    float* R2r     = (float*)(ws + 73802880);                 //     18,432 B
    float* v1b     = (float*)(ws + 73821312);                 //  2,359,296 B
    float* v2b     = (float*)(ws + 76180608);                 //  1,179,648 B
    float* outp    = (float*)d_out;

    // weight prep (transposed pack: [kk][oc][ic])
    pack_w_kernel<<<(9 * 512 * 256 + 255) / 256, 256, 0, stream>>>(W1, Wp1, 256, 512);
    pack_w_kernel<<<(9 * 256 * 128 + 255) / 256, 256, 0, stream>>>(W2, Wp2, 128, 256);
    prep_r1_kernel<<<2, 256, 0, stream>>>(R1, R1r);
    prep_r2_kernel<<<(16 * 32 * 9 + 255) / 256, 256, 0, stream>>>(R2, R2r);

    // conv1: pair(512) -> 256, WMMA bf16.  10 imgs * 96 rows * 4 ocsegs
    conv3x3_wmma<512, 4, true><<<10 * 96 * 4, 128, 0, stream>>>(
        focal, nullptr, Wp1, b1, O1);
    // conv2: 256 -> 128
    conv3x3_wmma<256, 2, false><<<10 * 96 * 2, 128, 0, stream>>>(
        nullptr, O1, Wp2, b2, O2);

    // tail
    cost_mean_kernel<<<(2 * 9216 + 255) / 256, 256, 0, stream>>>(O2, W3, b3, costagg);
    v1_kernel<<<(2 * 32 * 9216 + 255) / 256, 256, 0, stream>>>(costagg, R1r, rb1, v1b);
    v2_kernel<<<(2 * 16 * 9216 + 255) / 256, 256, 0, stream>>>(v1b, R2r, rb2, v2b);
    softmax_kernel<<<(2 * 9216) / 4, 128, 0, stream>>>(v2b, R3, rb3, outp);
}
